// PAGTNGNN_48095043780819
// MI455X (gfx1250) — compile-verified, matrix operand order checked by reading, not run on hardware
//
#include <hip/hip_runtime.h>

#define N_NODES 50000
#define N_EDGES 1600000
#define DIN     128
#define DD      64
#define HH      4
#define FEAT    16
#define NLAY    2
#define NH      (N_NODES * HH)           // 200,000
#define NHD     (N_NODES * HH * DD)      // 12,800,000

typedef __attribute__((ext_vector_type(16))) __bf16 v16bf;
typedef __attribute__((ext_vector_type(8)))  float  v8f;

// ---------------------------------------------------------------------------
// WMMA fragment helpers (wave32, 16x16x32 bf16 -> f32)
// A (16x32, 16-bit) layout: lanes 0-15 hold M=0-15, lanes 16-31 hold M=0-15;
//   lane half=lane>>4: vector elems 0..7 = K (half*8 + 0..7), elems 8..15 = K (16 + half*8 + 0..7)
// B (32x16) mirrors A with lane = column index.
// C/D f32 16x16: VGPR r: lanes0-15 -> (M=r, N=lane), lanes16-31 -> (M=8+r, N=lane-16)
// ---------------------------------------------------------------------------
__device__ inline v16bf load_a_frag(const float* __restrict__ A, int lda, int m0, int k0, int lane) {
  int half = (lane >> 4) & 1;
  int mr   = lane & 15;
  const float* p = A + (size_t)(m0 + mr) * lda + k0 + half * 8;
  v16bf a;
#pragma unroll
  for (int j = 0; j < 8; ++j) a[j] = (__bf16)p[j];
#pragma unroll
  for (int j = 0; j < 8; ++j) a[8 + j] = (__bf16)p[16 + j];
  return a;
}

// Pre-swizzle a (K x Nc) f32 row-major weight into bf16 B-fragments.
// frag layout: [kt][nt][lane][16] contiguous  (idx*16 below).
__global__ __launch_bounds__(256) void prep_wfrag_kernel(
    const float* __restrict__ W, __bf16* __restrict__ frag, int K, int Nc) {
  int idx = blockIdx.x * blockDim.x + threadIdx.x;
  int ktiles = K >> 5, ntiles = Nc >> 4;
  int total = ktiles * ntiles * 32;
  if (idx >= total) return;
  int lane = idx & 31;
  int t    = idx >> 5;
  int nt   = t % ntiles;
  int kt   = t / ntiles;
  int half = (lane >> 4) & 1;
  int n    = nt * 16 + (lane & 15);
  __bf16* o = frag + (size_t)idx * 16;
#pragma unroll
  for (int j = 0; j < 8; ++j) {
    o[j]     = (__bf16)W[(size_t)(kt * 32 + half * 8 + j) * Nc + n];
    o[8 + j] = (__bf16)W[(size_t)(kt * 32 + 16 + half * 8 + j) * Nc + n];
  }
}

// Input projection: atom_h = relu(X(50000x128) @ W(128x256) + b); also copy to atom_input.
__global__ __launch_bounds__(256) void gemm_inp_kernel(
    const float* __restrict__ X, const __bf16* __restrict__ frag,
    const float* __restrict__ bias, float* __restrict__ outA, float* __restrict__ outB) {
  int wid  = (blockIdx.x * blockDim.x + threadIdx.x) >> 5;
  int lane = threadIdx.x & 31;
  if (wid >= N_NODES / 16) return;          // wave-uniform
  int m0 = wid * 16;
  v16bf a[4];
#pragma unroll
  for (int kt = 0; kt < 4; ++kt) a[kt] = load_a_frag(X, DIN, m0, kt * 32, lane);
  int half = (lane >> 4) & 1, nl = lane & 15;
  for (int nt = 0; nt < 16; ++nt) {
    v8f c = {};
#pragma unroll
    for (int kt = 0; kt < 4; ++kt) {
      v16bf b = *(const v16bf*)(frag + ((size_t)(kt * 16 + nt) * 32 + lane) * 16);
      c = __builtin_amdgcn_wmma_f32_16x16x32_bf16(false, a[kt], false, b, (short)0, c, false, false);
    }
    float bv = bias[nt * 16 + nl];
    int coli = nt * 16 + nl;
#pragma unroll
    for (int r = 0; r < 8; ++r) {
      float v = c[r] + bv;
      v = v > 0.f ? v : 0.f;
      size_t o = (size_t)(m0 + half * 8 + r) * (HH * DD) + coli;
      outA[o] = v;
      outB[o] = v;
    }
  }
}

// Two projections sharing the same A: out{1,2} = A(200000x64) @ W{1,2}(64x64) + b{1,2}
__global__ __launch_bounds__(256) void gemm_proj2_kernel(
    const float* __restrict__ A,
    const __bf16* __restrict__ f1, const float* __restrict__ b1,
    const __bf16* __restrict__ f2, const float* __restrict__ b2,
    float* __restrict__ out1, float* __restrict__ out2) {
  int wid  = (blockIdx.x * blockDim.x + threadIdx.x) >> 5;
  int lane = threadIdx.x & 31;
  if (wid >= NH / 16) return;               // wave-uniform
  int m0 = wid * 16;
  v16bf a[2];
#pragma unroll
  for (int kt = 0; kt < 2; ++kt) a[kt] = load_a_frag(A, DD, m0, kt * 32, lane);
  int half = (lane >> 4) & 1, nl = lane & 15;
#pragma unroll
  for (int w = 0; w < 2; ++w) {
    const __bf16* f  = w ? f2 : f1;
    const float*  bb = w ? b2 : b1;
    float*        o  = w ? out2 : out1;
    for (int nt = 0; nt < 4; ++nt) {
      v8f c = {};
#pragma unroll
      for (int kt = 0; kt < 2; ++kt) {
        v16bf b = *(const v16bf*)(f + ((size_t)(kt * 4 + nt) * 32 + lane) * 16);
        c = __builtin_amdgcn_wmma_f32_16x16x32_bf16(false, a[kt], false, b, (short)0, c, false, false);
      }
      float bv = bb[nt * 16 + nl];
      int coli = nt * 16 + nl;
#pragma unroll
      for (int r = 0; r < 8; ++r)
        o[(size_t)(m0 + half * 8 + r) * DD + coli] = c[r] + bv;
    }
  }
}

// float atomic max via sign-aware int atomics (always compiles, lowers to HW atomics)
__device__ inline void atomicMaxF32(float* addr, float val) {
  if (val >= 0.f) atomicMax((int*)addr, __float_as_int(val));
  else            atomicMin((unsigned int*)addr, __float_as_uint(val));
}

__global__ __launch_bounds__(256) void init_layer_kernel(
    float* __restrict__ segmax, float* __restrict__ segsum, float* __restrict__ aggr) {
  int i = blockIdx.x * blockDim.x + threadIdx.x;
  if (i < NH) { segmax[i] = -__builtin_inff(); segsum[i] = 0.f; }
  if (i < NHD) aggr[i] = 0.f;
}

// One wave per (edge, head): logit = dot(lrelu(xs[col]+xd[row]+eattr@We+be), dotw)+dotb
__global__ __launch_bounds__(256) void edge_attn_kernel(
    const float* __restrict__ xs, const float* __restrict__ xd,
    const float* __restrict__ eattr,
    const float* __restrict__ Wedg, const float* __restrict__ bedg,
    const float* __restrict__ dotw, const float* __restrict__ dotb,
    const long long* __restrict__ row, const long long* __restrict__ col,
    float* __restrict__ alpha, float* __restrict__ segmax) {
  __shared__ float sW[FEAT * DD];
  __shared__ float sb[DD];
  for (int i = threadIdx.x; i < FEAT * DD; i += blockDim.x) sW[i] = Wedg[i];
  for (int i = threadIdx.x; i < DD; i += blockDim.x) sb[i] = bedg[i];
  __syncthreads();
  int wid  = (blockIdx.x * blockDim.x + threadIdx.x) >> 5;
  int lane = threadIdx.x & 31;
  int e = wid >> 2, h = wid & 3;
  if (e >= N_EDGES) return;
  int r = (int)row[e], c = (int)col[e];
  float eav = eattr[(size_t)e * FEAT + (lane & 15)];
  float ea0 = sb[lane], ea1 = sb[lane + 32];
#pragma unroll
  for (int f = 0; f < FEAT; ++f) {
    float w = __shfl(eav, f, 16);
    ea0 = fmaf(w, sW[f * DD + lane], ea0);
    ea1 = fmaf(w, sW[f * DD + lane + 32], ea1);
  }
  size_t bs = ((size_t)c * HH + h) * DD;
  size_t bd = ((size_t)r * HH + h) * DD;
  float v0 = xs[bs + lane]      + xd[bd + lane]      + ea0;
  float v1 = xs[bs + lane + 32] + xd[bd + lane + 32] + ea1;
  v0 = v0 > 0.f ? v0 : 0.2f * v0;
  v1 = v1 > 0.f ? v1 : 0.2f * v1;
  float acc = fmaf(v0, dotw[lane], v1 * dotw[lane + 32]);
#pragma unroll
  for (int off = 16; off >= 1; off >>= 1) acc += __shfl_xor(acc, off, 32);
  if (lane == 0) {
    float logit = acc + dotb[0];
    alpha[(size_t)e * HH + h] = logit;
    atomicMaxF32(&segmax[r * HH + h], logit);
  }
}

__global__ __launch_bounds__(256) void edge_expsum_kernel(
    const long long* __restrict__ row, const float* __restrict__ segmax,
    float* __restrict__ alpha, float* __restrict__ segsum) {
  int i = blockIdx.x * blockDim.x + threadIdx.x;
  if (i >= N_EDGES * HH) return;
  int e = i >> 2, h = i & 3;
  int r = (int)row[e];
  float m = segmax[r * HH + h];
  if (m == -__builtin_inff()) m = 0.f;     // match reference's finite-guard
  float ex = __expf(alpha[i] - m);
  alpha[i] = ex;
  atomicAdd(&segsum[r * HH + h], ex);
}

// One wave per (edge, head): scatter alpha*(xm[col]+eattr@Wm+bm) into aggr[col]
__global__ __launch_bounds__(256) void edge_msg_kernel(
    const float* __restrict__ xm, const float* __restrict__ eattr,
    const float* __restrict__ Wedg, const float* __restrict__ bedg,
    const long long* __restrict__ row, const long long* __restrict__ col,
    const float* __restrict__ alpha, const float* __restrict__ segsum,
    float* __restrict__ aggr) {
  __shared__ float sW[FEAT * DD];
  __shared__ float sb[DD];
  for (int i = threadIdx.x; i < FEAT * DD; i += blockDim.x) sW[i] = Wedg[i];
  for (int i = threadIdx.x; i < DD; i += blockDim.x) sb[i] = bedg[i];
  __syncthreads();
  int wid  = (blockIdx.x * blockDim.x + threadIdx.x) >> 5;
  int lane = threadIdx.x & 31;
  int e = wid >> 2, h = wid & 3;
  if (e >= N_EDGES) return;
  int r = (int)row[e], c = (int)col[e];
  float a = alpha[(size_t)e * HH + h] / (segsum[r * HH + h] + 1e-16f);
  float eav = eattr[(size_t)e * FEAT + (lane & 15)];
  float em0 = sb[lane], em1 = sb[lane + 32];
#pragma unroll
  for (int f = 0; f < FEAT; ++f) {
    float w = __shfl(eav, f, 16);
    em0 = fmaf(w, sW[f * DD + lane], em0);
    em1 = fmaf(w, sW[f * DD + lane + 32], em1);
  }
  size_t bm = ((size_t)c * HH + h) * DD;
  atomicAdd(&aggr[bm + lane],      a * (xm[bm + lane]      + em0));
  atomicAdd(&aggr[bm + lane + 32], a * (xm[bm + lane + 32] + em1));
}

__global__ __launch_bounds__(256) void combine_kernel(
    const float* __restrict__ aggr, const float* __restrict__ xw,
    const float* __restrict__ atom_input, float* __restrict__ atom_h) {
  int i = blockIdx.x * blockDim.x + threadIdx.x;
  if (i >= NHD) return;
  float v = aggr[i] + xw[i] + atom_input[i];
  atom_h[i] = v > 0.f ? v : 0.f;
}

__global__ __launch_bounds__(256) void mean_kernel(
    const float* __restrict__ atom_h, float* __restrict__ out) {
  int i = blockIdx.x * blockDim.x + threadIdx.x;
  if (i >= N_NODES * DD) return;
  int n = i / DD, d = i % DD;
  float s = 0.f;
#pragma unroll
  for (int h = 0; h < HH; ++h) s += atom_h[((size_t)n * HH + h) * DD + d];
  out[i] = 0.25f * s;
}

extern "C" void kernel_launch(void* const* d_in, const int* in_sizes, int n_in,
                              void* d_out, int out_size, void* d_ws, size_t ws_size,
                              hipStream_t stream) {
  (void)in_sizes; (void)n_in; (void)out_size; (void)ws_size;
  const float*     x          = (const float*)d_in[0];
  const float*     edge_attr  = (const float*)d_in[1];
  const long long* ei         = (const long long*)d_in[2];
  const long long* row        = ei;              // edge_index[0]
  const long long* col        = ei + N_EDGES;    // edge_index[1]
  const float* atom_inp_w = (const float*)d_in[3];
  const float* atom_inp_b = (const float*)d_in[4];
  const float* attn_src_w = (const float*)d_in[5];
  const float* attn_src_b = (const float*)d_in[6];
  const float* attn_dst_w = (const float*)d_in[7];
  const float* attn_dst_b = (const float*)d_in[8];
  const float* attn_edg_w = (const float*)d_in[9];
  const float* attn_edg_b = (const float*)d_in[10];
  const float* attn_dot_w = (const float*)d_in[11];
  const float* attn_dot_b = (const float*)d_in[12];
  const float* msg_dst_w  = (const float*)d_in[13];
  const float* msg_dst_b  = (const float*)d_in[14];
  const float* msg_edg_w  = (const float*)d_in[15];
  const float* msg_edg_b  = (const float*)d_in[16];
  const float* wgt_n_w    = (const float*)d_in[17];
  const float* wgt_n_b    = (const float*)d_in[18];
  float* out = (float*)d_out;

  char* ws = (char*)d_ws;
  size_t off = 0;
  auto alloc = [&](size_t bytes) -> void* {
    void* p = ws + off;
    off += (bytes + 255) & ~(size_t)255;
    return p;
  };
  float*  atom_h     = (float*)alloc((size_t)NHD * 4);
  float*  atom_input = (float*)alloc((size_t)NHD * 4);
  float*  bufA       = (float*)alloc((size_t)NHD * 4);   // xs, later xm
  float*  bufB       = (float*)alloc((size_t)NHD * 4);   // xd, later xw
  float*  aggr       = (float*)alloc((size_t)NHD * 4);
  float*  alpha      = (float*)alloc((size_t)N_EDGES * HH * 4);
  float*  segmax     = (float*)alloc((size_t)NH * 4);
  float*  segsum     = (float*)alloc((size_t)NH * 4);
  __bf16* frag_inp   = (__bf16*)alloc((size_t)DIN * HH * DD * 2);
  __bf16* frag_lw    = (__bf16*)alloc((size_t)NLAY * 4 * DD * DD * 2);

  // --- weight pre-swizzle into WMMA B-fragment layout (bf16) ---
  prep_wfrag_kernel<<<(4 * 16 * 32 + 255) / 256, 256, 0, stream>>>(atom_inp_w, frag_inp, DIN, HH * DD);
  for (int l = 0; l < NLAY; ++l) {
    prep_wfrag_kernel<<<1, 256, 0, stream>>>(attn_src_w + l * DD * DD, frag_lw + (size_t)(l * 4 + 0) * 4096, DD, DD);
    prep_wfrag_kernel<<<1, 256, 0, stream>>>(attn_dst_w + l * DD * DD, frag_lw + (size_t)(l * 4 + 1) * 4096, DD, DD);
    prep_wfrag_kernel<<<1, 256, 0, stream>>>(msg_dst_w  + l * DD * DD, frag_lw + (size_t)(l * 4 + 2) * 4096, DD, DD);
    prep_wfrag_kernel<<<1, 256, 0, stream>>>(wgt_n_w    + l * DD * DD, frag_lw + (size_t)(l * 4 + 3) * 4096, DD, DD);
  }

  // --- input projection (WMMA): atom_h = atom_input = relu(x @ W + b) ---
  gemm_inp_kernel<<<(N_NODES / 16 + 7) / 8, 256, 0, stream>>>(x, frag_inp, atom_inp_b, atom_h, atom_input);

  for (int l = 0; l < NLAY; ++l) {
    init_layer_kernel<<<(NHD + 255) / 256, 256, 0, stream>>>(segmax, segsum, aggr);
    // attention projections (WMMA)
    gemm_proj2_kernel<<<(NH / 16 + 7) / 8, 256, 0, stream>>>(
        atom_h,
        frag_lw + (size_t)(l * 4 + 0) * 4096, attn_src_b + l * DD,
        frag_lw + (size_t)(l * 4 + 1) * 4096, attn_dst_b + l * DD,
        bufA, bufB);
    // edge logits + segment max
    edge_attn_kernel<<<N_EDGES * HH / 8, 256, 0, stream>>>(
        bufA, bufB, edge_attr,
        attn_edg_w + l * FEAT * DD, attn_edg_b + l * DD,
        attn_dot_w + l * DD, attn_dot_b + l,
        row, col, alpha, segmax);
    // softmax exp + segment sum
    edge_expsum_kernel<<<(N_EDGES * HH + 255) / 256, 256, 0, stream>>>(row, segmax, alpha, segsum);
    // message projections (WMMA)
    gemm_proj2_kernel<<<(NH / 16 + 7) / 8, 256, 0, stream>>>(
        atom_h,
        frag_lw + (size_t)(l * 4 + 2) * 4096, msg_dst_b + l * DD,
        frag_lw + (size_t)(l * 4 + 3) * 4096, wgt_n_b + l * DD,
        bufA, bufB);
    // weighted message scatter
    edge_msg_kernel<<<N_EDGES * HH / 8, 256, 0, stream>>>(
        bufA, edge_attr,
        msg_edg_w + l * FEAT * DD, msg_edg_b + l * DD,
        row, col, alpha, segsum, aggr);
    // residual + relu
    combine_kernel<<<(NHD + 255) / 256, 256, 0, stream>>>(aggr, bufB, atom_input, atom_h);
  }
  mean_kernel<<<(N_NODES * DD + 255) / 256, 256, 0, stream>>>(atom_h, out);
}
